// THU_NGN_11003706212754
// MI455X (gfx1250) — matrix operand order repaired and checked by reading, not compiled, not run on hardware
//
#include <hip/hip_runtime.h>
#include <hip/hip_bf16.h>
#include <math.h>

typedef __bf16 bf16;
typedef __attribute__((ext_vector_type(16))) __bf16 v16bf;
typedef __attribute__((ext_vector_type(8)))  float  v8f;

#define DEVINL __device__ __forceinline__

// ---------------------------------------------------------------------------
// WMMA helpers (CDNA5 wave32, 16x16x32 bf16 -> f32)
// ---------------------------------------------------------------------------
DEVINL v8f wmma_bf(v16bf a, v16bf b, v8f c) {
  return __builtin_amdgcn_wmma_f32_16x16x32_bf16(false, a, false, b, (short)0, c,
                                                 false, false);
}

DEVINL float sigf(float x) { return 1.0f / (1.0f + __expf(-x)); }

// A tile 16x32 (MxK), row-major source. Lanes 0-15 hold K 0-15 in
// elems 0-15; lanes 16-31 hold K 16-31.
DEVINL v16bf load_a_tile(const bf16* __restrict__ A, size_t lda, int lane, int koff) {
  int m = lane & 15, half = lane >> 4;
  const bf16* p = A + (size_t)m * lda + (size_t)koff + 16 * half;
  v16bf a;
#pragma unroll
  for (int i = 0; i < 8; ++i) a[i] = p[i];
#pragma unroll
  for (int i = 8; i < 16; ++i) a[i] = p[i];
  return a;
}

// B tile 32x16 (KxN) where memory holds B^T: W[n][k] row-major (ld = ldw).
// Lanes 0-15: K 0-15 for column n=lane; lanes 16-31: K 16-31.
DEVINL v16bf load_bT_tile(const bf16* __restrict__ W, size_t ldw, int lane, int koff) {
  int n = lane & 15, half = lane >> 4;
  const bf16* p = W + (size_t)n * ldw + (size_t)koff + 16 * half;
  v16bf b;
#pragma unroll
  for (int i = 0; i < 16; ++i) b[i] = p[i];
  return b;
}

// ---------------------------------------------------------------------------
// Utility kernels
// ---------------------------------------------------------------------------
__global__ void cvt_f32_bf16_pad(const float* __restrict__ src, bf16* __restrict__ dst,
                                 int rows, int cols, int colsPad) {
  size_t idx = (size_t)blockIdx.x * blockDim.x + threadIdx.x;
  size_t total = (size_t)rows * colsPad;
  if (idx >= total) return;
  int r = (int)(idx / colsPad);
  int c = (int)(idx % colsPad);
  dst[idx] = (c < cols) ? (bf16)src[(size_t)r * cols + c] : (bf16)0.0f;
}

// conv weights [F][C][k] f32 -> wT[dk][F][C] bf16 (C contiguous per row)
__global__ void cvt_convw_kernel(const float* __restrict__ src, bf16* __restrict__ dst,
                                 int k) {
  size_t idx = (size_t)blockIdx.x * blockDim.x + threadIdx.x;
  size_t total = (size_t)128 * 512 * k;
  if (idx >= total) return;
  int f = (int)(idx / (512 * k));
  int rem = (int)(idx % (512 * k));
  int c = rem / k;
  int dk = rem % k;
  dst[((size_t)dk * 128 + f) * 512 + c] = (bf16)src[idx];
}

__global__ void zero_u32_kernel(unsigned int* __restrict__ p, size_t n) {
  size_t i = (size_t)blockIdx.x * blockDim.x + threadIdx.x;
  if (i < n) p[i] = 0u;
}

__global__ void add_bias_kernel(const float* __restrict__ a, const float* __restrict__ b,
                                float* __restrict__ o, int n) {
  int i = blockIdx.x * blockDim.x + threadIdx.x;
  if (i < n) o[i] = a[i] + b[i];
}

// ---------------------------------------------------------------------------
// Embedding gather -> bf16, forward + per-sequence-reversed copies.
// Row layout: row = t*64 + b, K padded 300 -> 320.
// ---------------------------------------------------------------------------
__global__ void embed_kernel(const int* __restrict__ tokens, const int* __restrict__ lengths,
                             const float* __restrict__ emb,
                             bf16* __restrict__ Xf, bf16* __restrict__ Xr) {
  int row = blockIdx.x;            // 0 .. 32767
  int t = row >> 6, b = row & 63;
  int L = lengths[b];
  int rt = (t < L) ? (L - 1 - t) : t;
  int tokF = tokens[b * 512 + t];
  int tokR = tokens[b * 512 + rt];
  const float* ef = emb + (size_t)tokF * 300;
  const float* er = emb + (size_t)tokR * 300;
  bf16* xf = Xf + (size_t)row * 320;
  bf16* xr = Xr + (size_t)row * 320;
  for (int e = threadIdx.x; e < 320; e += 64) {
    xf[e] = (e < 300) ? (bf16)ef[e] : (bf16)0.0f;
    xr[e] = (e < 300) ? (bf16)er[e] : (bf16)0.0f;
  }
}

// ---------------------------------------------------------------------------
// Input projection GEMM: Y[32768 x 1024] = X[32768 x 320] * W^T + bias
// Each wave: one 16-row M tile x 32-wide N strip (two 16x16 WMMA tiles).
// ---------------------------------------------------------------------------
__global__ void gemm_xproj_kernel(const bf16* __restrict__ X, const bf16* __restrict__ W,
                                  const float* __restrict__ bias, float* __restrict__ Y) {
  const int KP = 320;
  int tid = threadIdx.x, wid = tid >> 5, lane = tid & 31;
  int wgid = blockIdx.x * 8 + wid;      // 65536 wide tiles = 2048 Mtiles * 32 Nstrips
  int mt = wgid >> 5;
  int n0 = (wgid & 31) * 32;
  int n = lane & 15, half = lane >> 4;

  const bf16* Abase = X + (size_t)(mt * 16) * KP;
  const bf16* W0 = W + (size_t)n0 * KP;
  const bf16* W1 = W + (size_t)(n0 + 16) * KP;

  v8f acc0 = {};
  v8f acc1 = {};
#pragma unroll
  for (int kk = 0; kk < 10; ++kk) {
    __builtin_prefetch(Abase + (size_t)(lane & 15) * KP + kk * 32 + 256, 0, 3);
    v16bf a = load_a_tile(Abase, KP, lane, kk * 32);
    v16bf b0 = load_bT_tile(W0, KP, lane, kk * 32);
    v16bf b1 = load_bT_tile(W1, KP, lane, kk * 32);
    acc0 = wmma_bf(a, b0, acc0);
    acc1 = wmma_bf(a, b1, acc1);
  }
#pragma unroll
  for (int r = 0; r < 8; ++r) {
    int m = r + 8 * half;
    size_t ro = (size_t)(mt * 16 + m) * 1024;
    Y[ro + n0 + n]      = acc0[r] + bias[n0 + n];
    Y[ro + n0 + 16 + n] = acc1[r] + bias[n0 + 16 + n];
  }
}

// ---------------------------------------------------------------------------
// One LSTM timestep: g = xg[t] + h_in @ Whh^T  (WMMA), then cell update.
// 8 workgroups, each owns hidden slice j in [32*blk, 32*blk+32) for all 4 gates,
// so i/f/g/o land in this WG's LDS and the nonlinearity fuses in.
// h double-buffered (bf16) to avoid cross-WG read/write races within a step.
// ---------------------------------------------------------------------------
__global__ void lstm_step_kernel(const float* __restrict__ xg_t, const bf16* __restrict__ Whh,
                                 const bf16* __restrict__ h_in, bf16* __restrict__ h_out,
                                 float* __restrict__ cstate, float* __restrict__ hcat,
                                 const int* __restrict__ lengths, int t, int dirofs,
                                 int is_bwd) {
  __shared__ float gbuf[4 * 64 * 32];    // [gate][b][jj]  32 KB
  int j0 = blockIdx.x * 32;
  int tid = threadIdx.x, wid = tid >> 5, lane = tid & 31;
  int gate = wid >> 1, ncol = wid & 1;
  int n = lane & 15, half = lane >> 4;
  int nglob = gate * 256 + j0 + ncol * 16;

  const bf16* Wb = Whh + (size_t)nglob * 256;

  for (int mtile = 0; mtile < 4; ++mtile) {
    v8f acc;
#pragma unroll
    for (int r = 0; r < 8; ++r) {
      int m = r + 8 * half;
      acc[r] = xg_t[(size_t)(mtile * 16 + m) * 1024 + nglob + n];
    }
    const bf16* Ab = h_in + (size_t)(mtile * 16) * 256;
#pragma unroll
    for (int kk = 0; kk < 8; ++kk) {
      v16bf a = load_a_tile(Ab, 256, lane, kk * 32);
      v16bf b = load_bT_tile(Wb, 256, lane, kk * 32);
      acc = wmma_bf(a, b, acc);
    }
#pragma unroll
    for (int r = 0; r < 8; ++r) {
      int m = r + 8 * half;
      int bb = mtile * 16 + m;
      gbuf[(gate * 64 + bb) * 32 + ncol * 16 + n] = acc[r];
    }
  }
  __syncthreads();

  for (int e2 = tid; e2 < 2048; e2 += 256) {
    int bb = e2 >> 5, jj = e2 & 31;
    int j = j0 + jj;
    float gi = gbuf[(0 * 64 + bb) * 32 + jj];
    float gf = gbuf[(1 * 64 + bb) * 32 + jj];
    float gg = gbuf[(2 * 64 + bb) * 32 + jj];
    float go = gbuf[(3 * 64 + bb) * 32 + jj];
    float cprev = cstate[bb * 256 + j];
    float cc = sigf(gf) * cprev + sigf(gi) * tanhf(gg);
    float hh = sigf(go) * tanhf(cc);
    cstate[bb * 256 + j] = cc;
    int trow = t;
    if (is_bwd) { int L = lengths[bb]; trow = (t < L) ? (L - 1 - t) : t; }
    hcat[(size_t)(trow * 64 + bb) * 512 + dirofs + j] = hh;
    h_out[bb * 256 + j] = (bf16)hh;
  }
}

// ---------------------------------------------------------------------------
// Attention: e[t,b] = exp(tanh(h).attn_w + b) * mask. One wave per (t,b).
// ---------------------------------------------------------------------------
__global__ void attn_e_kernel(const float* __restrict__ hcat, const float* __restrict__ attn_w,
                              const float* __restrict__ attn_b, const int* __restrict__ lengths,
                              float* __restrict__ e) {
  int tid = threadIdx.x, wid = tid >> 5, lane = tid & 31;
  int gid = blockIdx.x * 8 + wid;        // t*64 + b
  int t = gid >> 6, b = gid & 63;
  const float* hp = hcat + (size_t)gid * 512;
  float s = 0.0f;
#pragma unroll
  for (int i = 0; i < 16; ++i) {
    int c = lane * 16 + i;
    s += tanhf(hp[c]) * attn_w[c];
  }
#pragma unroll
  for (int m = 16; m >= 1; m >>= 1) s += __shfl_xor(s, m, 32);
  if (lane == 0) {
    float u = s + attn_b[0];
    e[gid] = (t < lengths[b]) ? __expf(u) : 0.0f;
  }
}

__global__ void sum_e_kernel(const float* __restrict__ e, float* __restrict__ inv) {
  int b = threadIdx.x;
  if (b < 64) {
    float s = 0.0f;
    for (int t = 0; t < 512; ++t) s += e[t * 64 + b];
    inv[b] = 1.0f / s;
  }
}

// rT[b][t][c] bf16, t padded to 528 (zero rows), c contiguous for conv K-dim.
__global__ void scale_r_kernel(const float* __restrict__ hcat, const float* __restrict__ e,
                               const float* __restrict__ inv, bf16* __restrict__ rT) {
  const int Tp = 528, C = 512;
  size_t idx = (size_t)blockIdx.x * blockDim.x + threadIdx.x;
  size_t total = (size_t)64 * Tp * C;
  if (idx >= total) return;
  int c = (int)(idx % C);
  size_t rem = idx / C;
  int t = (int)(rem % Tp);
  int b = (int)(rem / Tp);
  float v = 0.0f;
  if (t < 512) {
    v = hcat[(size_t)(t * 64 + b) * 512 + c] * e[t * 64 + b] * inv[b];
  }
  rT[idx] = (bf16)v;
}

// ---------------------------------------------------------------------------
// Conv1d(k) + global max pool, implicit-GEMM via WMMA.
//   y[f, t] = sum_dk ( W_dk[F,C] @ rT[t+dk][C] )   -- all K loads contiguous.
// Grid: 64 batches x 8 f-tiles. Each wave sweeps 16-wide time tiles.
// ---------------------------------------------------------------------------
__global__ void conv_pool_kernel(const bf16* __restrict__ rT, const bf16* __restrict__ wT,
                                 const float* __restrict__ bias, float* __restrict__ cat,
                                 int k, int Tout, int catofs) {
  const int C = 512, Tp = 528;
  __shared__ float wmax[8][16];
  int bb = blockIdx.x >> 3;
  int f0 = (blockIdx.x & 7) * 16;
  int tid = threadIdx.x, wid = tid >> 5, lane = tid & 31;
  int n = lane & 15, half = lane >> 4;
  int NT = (Tout + 15) >> 4;
  const bf16* rb = rT + (size_t)bb * Tp * C;

  float rmax[8];
#pragma unroll
  for (int r = 0; r < 8; ++r) rmax[r] = -1e30f;

  for (int tt = wid; tt < NT; tt += 8) {
    int t0 = tt * 16;
    v8f acc = {};
    for (int dk = 0; dk < k; ++dk) {
      const bf16* Ab = wT + ((size_t)dk * 128 + f0) * C;   // wT[dk][f][c]
      const bf16* Bb = rb + (size_t)(t0 + dk) * C;         // rows = time
#pragma unroll
      for (int kk = 0; kk < 16; ++kk) {
        v16bf a = load_a_tile(Ab, C, lane, kk * 32);
        v16bf b = load_bT_tile(Bb, C, lane, kk * 32);
        acc = wmma_bf(a, b, acc);
      }
    }
    bool valid = (t0 + n) < Tout;
#pragma unroll
    for (int r = 0; r < 8; ++r) rmax[r] = fmaxf(rmax[r], valid ? acc[r] : -1e30f);
  }
  // reduce over the 16 lanes of each half (N dimension of the D tile)
#pragma unroll
  for (int r = 0; r < 8; ++r) {
    float v = rmax[r];
#pragma unroll
    for (int m = 1; m <= 8; m <<= 1) v = fmaxf(v, __shfl_xor(v, m, 32));
    rmax[r] = v;
  }
  if (n == 0) {
#pragma unroll
    for (int r = 0; r < 8; ++r) wmax[wid][r + 8 * half] = rmax[r];
  }
  __syncthreads();
  if (tid < 16) {
    float v = wmax[0][tid];
#pragma unroll
    for (int wv = 1; wv < 8; ++wv) v = fmaxf(v, wmax[wv][tid]);
    cat[(size_t)bb * 512 + catofs + f0 + tid] = v + bias[f0 + tid];
  }
}

// ---------------------------------------------------------------------------
// FC1: out[64x256] = relu(cat[64x512] @ fc1_w^T + b). 64 WMMA tiles, 8 WGs.
// ---------------------------------------------------------------------------
__global__ void fc1_kernel(const bf16* __restrict__ catb, const bf16* __restrict__ w,
                           const float* __restrict__ bias, float* __restrict__ out) {
  int tid = threadIdx.x, wid = tid >> 5, lane = tid & 31;
  int tile = blockIdx.x * 8 + wid;       // 0..63
  int mt = tile & 3, nt = tile >> 2;
  int n = lane & 15, half = lane >> 4;
  const bf16* Ab = catb + (size_t)(mt * 16) * 512;
  const bf16* Wb = w + (size_t)(nt * 16) * 512;
  v8f acc = {};
#pragma unroll
  for (int kk = 0; kk < 16; ++kk) {
    v16bf a = load_a_tile(Ab, 512, lane, kk * 32);
    v16bf b = load_bT_tile(Wb, 512, lane, kk * 32);
    acc = wmma_bf(a, b, acc);
  }
#pragma unroll
  for (int r = 0; r < 8; ++r) {
    int m = r + 8 * half;
    float v = acc[r] + bias[nt * 16 + n];
    out[(size_t)(mt * 16 + m) * 256 + nt * 16 + n] = (v > 0.0f) ? v : 0.0f;
  }
}

__global__ void fc2_kernel(const float* __restrict__ fin, const float* __restrict__ w,
                           const float* __restrict__ b, float* __restrict__ out) {
  int tid = threadIdx.x;
  if (tid < 64) {
    float s = 0.0f;
    for (int d = 0; d < 256; ++d) s += fin[tid * 256 + d] * w[d];
    s += b[0];
    out[tid] = 1.0f / (1.0f + __expf(-s));
  }
}

// ---------------------------------------------------------------------------
// Host launcher
// ---------------------------------------------------------------------------
extern "C" void kernel_launch(void* const* d_in, const int* in_sizes, int n_in,
                              void* d_out, int out_size, void* d_ws, size_t ws_size,
                              hipStream_t stream) {
  (void)in_sizes; (void)n_in; (void)out_size; (void)ws_size;

  const int* tokens   = (const int*)d_in[0];
  const int* lengths  = (const int*)d_in[1];
  const float* emb    = (const float*)d_in[2];
  const float* Wih_f  = (const float*)d_in[3];
  const float* Whh_f  = (const float*)d_in[4];
  const float* bih_f  = (const float*)d_in[5];
  const float* bhh_f  = (const float*)d_in[6];
  const float* Wih_b  = (const float*)d_in[7];
  const float* Whh_b  = (const float*)d_in[8];
  const float* bih_b  = (const float*)d_in[9];
  const float* bhh_b  = (const float*)d_in[10];
  const float* attn_w = (const float*)d_in[11];
  const float* attn_b = (const float*)d_in[12];
  const float* conv_w[4] = { (const float*)d_in[13], (const float*)d_in[15],
                             (const float*)d_in[17], (const float*)d_in[19] };
  const float* conv_b[4] = { (const float*)d_in[14], (const float*)d_in[16],
                             (const float*)d_in[18], (const float*)d_in[20] };
  const float* fc1_w  = (const float*)d_in[21];
  const float* fc1_b  = (const float*)d_in[22];
  const float* fc2_w  = (const float*)d_in[23];
  const float* fc2_b  = (const float*)d_in[24];
  float* out = (float*)d_out;

  // ---- workspace carve-up ----
  char* base = (char*)d_ws;
  size_t off = 0;
  auto alloc = [&](size_t bytes) -> void* {
    void* p = base + off;
    off = (off + bytes + 255) & ~((size_t)255);
    return p;
  };
  float* xgF    = (float*)alloc((size_t)32768 * 1024 * 4);
  float* xgB    = (float*)alloc((size_t)32768 * 1024 * 4);
  float* hcat   = (float*)alloc((size_t)32768 * 512 * 4);
  float* cstate = (float*)alloc((size_t)64 * 256 * 4);
  float* biasF  = (float*)alloc(1024 * 4);
  float* biasB  = (float*)alloc(1024 * 4);
  float* ebuf   = (float*)alloc(32768 * 4);
  float* inv    = (float*)alloc(64 * 4);
  float* catf   = (float*)alloc((size_t)64 * 512 * 4);
  float* fc1out = (float*)alloc((size_t)64 * 256 * 4);
  bf16* Xf    = (bf16*)alloc((size_t)32768 * 320 * 2);
  bf16* Xr    = (bf16*)alloc((size_t)32768 * 320 * 2);
  bf16* wihF  = (bf16*)alloc((size_t)1024 * 320 * 2);
  bf16* wihB  = (bf16*)alloc((size_t)1024 * 320 * 2);
  bf16* whhF  = (bf16*)alloc((size_t)1024 * 256 * 2);
  bf16* whhB  = (bf16*)alloc((size_t)1024 * 256 * 2);
  bf16* wcb[4];
  const int ksz[4]  = {3, 5, 7, 9};
  const int tout[4] = {510, 508, 506, 504};
  for (int i = 0; i < 4; ++i) wcb[i] = (bf16*)alloc((size_t)128 * 512 * ksz[i] * 2);
  bf16* fc1wb = (bf16*)alloc((size_t)256 * 512 * 2);
  bf16* catbb = (bf16*)alloc((size_t)64 * 512 * 2);
  bf16* h0    = (bf16*)alloc((size_t)64 * 256 * 2);
  bf16* h1    = (bf16*)alloc((size_t)64 * 256 * 2);
  bf16* rT    = (bf16*)alloc((size_t)64 * 528 * 512 * 2);

  auto blk = [](size_t total, int t) { return (int)((total + t - 1) / t); };

  // ---- weight conversion to bf16 (once per call) ----
  cvt_f32_bf16_pad<<<blk((size_t)1024 * 320, 256), 256, 0, stream>>>(Wih_f, wihF, 1024, 300, 320);
  cvt_f32_bf16_pad<<<blk((size_t)1024 * 320, 256), 256, 0, stream>>>(Wih_b, wihB, 1024, 300, 320);
  cvt_f32_bf16_pad<<<blk((size_t)1024 * 256, 256), 256, 0, stream>>>(Whh_f, whhF, 1024, 256, 256);
  cvt_f32_bf16_pad<<<blk((size_t)1024 * 256, 256), 256, 0, stream>>>(Whh_b, whhB, 1024, 256, 256);
  for (int i = 0; i < 4; ++i) {
    cvt_convw_kernel<<<blk((size_t)128 * 512 * ksz[i], 256), 256, 0, stream>>>(conv_w[i],
                                                                               wcb[i], ksz[i]);
  }
  cvt_f32_bf16_pad<<<blk((size_t)256 * 512, 256), 256, 0, stream>>>(fc1_w, fc1wb, 256, 512, 512);
  add_bias_kernel<<<4, 256, 0, stream>>>(bih_f, bhh_f, biasF, 1024);
  add_bias_kernel<<<4, 256, 0, stream>>>(bih_b, bhh_b, biasB, 1024);

  // ---- embedding gather (fwd + reversed) ----
  embed_kernel<<<32768, 64, 0, stream>>>(tokens, lengths, emb, Xf, Xr);

  // ---- input projection GEMMs ----
  gemm_xproj_kernel<<<8192, 256, 0, stream>>>(Xf, wihF, biasF, xgF);
  gemm_xproj_kernel<<<8192, 256, 0, stream>>>(Xr, wihB, biasB, xgB);

  // ---- BiLSTM recurrence: 512 step-kernels per direction ----
  for (int dir = 0; dir < 2; ++dir) {
    const float* xg  = dir ? xgB : xgF;
    const bf16*  whh = dir ? whhB : whhF;
    int dirofs = dir ? 256 : 0;
    zero_u32_kernel<<<blk(8192, 256), 256, 0, stream>>>((unsigned int*)h0, 8192);
    zero_u32_kernel<<<blk(8192, 256), 256, 0, stream>>>((unsigned int*)h1, 8192);
    zero_u32_kernel<<<blk(16384, 256), 256, 0, stream>>>((unsigned int*)cstate, 16384);
    for (int t = 0; t < 512; ++t) {
      bf16* hin  = (t & 1) ? h1 : h0;
      bf16* hout = (t & 1) ? h0 : h1;
      lstm_step_kernel<<<8, 256, 0, stream>>>(xg + (size_t)t * 64 * 1024, whh, hin, hout,
                                              cstate, hcat, lengths, t, dirofs, dir);
    }
  }

  // ---- attention ----
  attn_e_kernel<<<4096, 256, 0, stream>>>(hcat, attn_w, attn_b, lengths, ebuf);
  sum_e_kernel<<<1, 64, 0, stream>>>(ebuf, inv);
  scale_r_kernel<<<blk((size_t)64 * 528 * 512, 256), 256, 0, stream>>>(hcat, ebuf, inv, rT);

  // ---- convs + max pool ----
  for (int i = 0; i < 4; ++i) {
    conv_pool_kernel<<<512, 256, 0, stream>>>(rT, wcb[i], conv_b[i], catf,
                                              ksz[i], tout[i], i * 128);
  }

  // ---- FC head ----
  cvt_f32_bf16_pad<<<blk((size_t)64 * 512, 256), 256, 0, stream>>>(catf, catbb, 64, 512, 512);
  fc1_kernel<<<8, 256, 0, stream>>>(catbb, fc1wb, fc1_b, fc1out);
  fc2_kernel<<<1, 64, 0, stream>>>(fc1out, fc2_w, fc2_b, out);
}
